// CausalSelfAttention_2422361555823
// MI455X (gfx1250) — matrix operand order repaired and checked by reading, not compiled, not run on hardware
//
#include <hip/hip_runtime.h>

#define B_  2
#define T_  2048
#define C_  1024
#define H_  16
#define HD_ 64
#define M_  (B_ * T_)   // 4096

typedef __attribute__((ext_vector_type(16))) __bf16 v16bf;
typedef __attribute__((ext_vector_type(8)))  __bf16 v8bf;
typedef __attribute__((ext_vector_type(8)))  float  v8f;
typedef __attribute__((ext_vector_type(4)))  int    i4v;
typedef __attribute__((ext_vector_type(8)))  int    i8v;
typedef __attribute__((ext_vector_type(4)))  unsigned u4v;

#if defined(__gfx1250__) && __has_builtin(__builtin_amdgcn_global_load_async_to_lds_b128)
#define HAS_ASYNC_LDS 1
#else
#define HAS_ASYNC_LDS 0
#endif

#if defined(__gfx1250__) && __has_builtin(__builtin_amdgcn_tensor_load_to_lds)
#define HAS_TDM 1
#else
#define HAS_TDM 0
#endif

typedef __attribute__((address_space(1))) i4v* gvec_ptr;   // global int4*
typedef __attribute__((address_space(3))) i4v* lvec_ptr;   // LDS int4*

__device__ __forceinline__ void async_cp16(const unsigned short* g, unsigned short* l) {
#if HAS_ASYNC_LDS
  __builtin_amdgcn_global_load_async_to_lds_b128(
      (gvec_ptr)(unsigned long long)g,
      (lvec_ptr)(unsigned)(unsigned long long)l, 0, 0);
#else
  *(uint4*)l = *(const uint4*)g;
#endif
}

__device__ __forceinline__ void wait_async0() {
#if HAS_ASYNC_LDS
#if __has_builtin(__builtin_amdgcn_s_wait_asynccnt)
  __builtin_amdgcn_s_wait_asynccnt(0);
#else
  asm volatile("s_wait_asynccnt 0" ::: "memory");
#endif
#endif
}

#if HAS_TDM
// TDM load of one K tile: 32 rows x 64 bf16 from [T][64] into padded LDS rows
// (stride 80 halves = 160B). Pad: every 32 DWORDs (code 4) insert 8 DWORDs (code 7).
__device__ __forceinline__ void tdm_load_k_tile(const unsigned short* gbase,
                                                unsigned lds_off) {
  unsigned long long ga = (unsigned long long)gbase;
  u4v g0;
  g0[0] = 1u;                                            // count=1, user mode
  g0[1] = lds_off;                                       // lds_addr
  g0[2] = (unsigned)ga;                                  // global_addr[31:0]
  g0[3] = ((unsigned)(ga >> 32) & 0x01FFFFFFu) | (2u << 30);  // addr[56:32] | type=2
  i8v g1;
  g1[0] = (1 << 16) | (1 << 20) | (4 << 22) | (7 << 25); // data_size=2B, pad_en, pad codes
  g1[1] = (int)(64u << 16);                              // tensor_dim0 = 64 (bits 79:48)
  g1[2] = (int)(2048u << 16);                            // tensor_dim1 = 2048 (bits 111:80)
  g1[3] = (int)(64u << 16);                              // tile_dim0 = 64 (bits 127:112)
  g1[4] = 32;                                            // tile_dim1 = 32, tile_dim2 = 0
  g1[5] = 64;                                            // tensor_dim0_stride = 64
  g1[6] = 0;
  g1[7] = 0;
  i4v z4 = {};
  i8v z8 = {};
  __builtin_amdgcn_tensor_load_to_lds(g0, g1, z4, z4, z8, 0);
}
__device__ __forceinline__ void wait_tensor0() {
  __builtin_amdgcn_s_wait_tensorcnt(0);
}
#endif

__device__ __forceinline__ unsigned short f2bf(float f) {
  unsigned u = __float_as_uint(f);
  return (unsigned short)((u + 0x7FFFu + ((u >> 16) & 1u)) >> 16);
}

__device__ __forceinline__ v16bf cat8(v8bf lo, v8bf hi) {
  return __builtin_shufflevector(lo, hi, 0,1,2,3,4,5,6,7,8,9,10,11,12,13,14,15);
}

__device__ __forceinline__ v8f v8f_zero() {
  v8f z = {};
  return z;
}

// ---------------- fp32 -> bf16 conversion ----------------
__global__ void cvt_f32_bf16(const float* __restrict__ src,
                             unsigned short* __restrict__ dst, int n4) {
  int i = blockIdx.x * blockDim.x + threadIdx.x;
  int stride = gridDim.x * blockDim.x;
  for (; i < n4; i += stride) {
    float4 v = ((const float4*)src)[i];
    ushort4 o;
    o.x = f2bf(v.x); o.y = f2bf(v.y); o.z = f2bf(v.z); o.w = f2bf(v.w);
    ((ushort4*)dst)[i] = o;
  }
}

// ---------------- tiled bf16 WMMA GEMM, double-buffered ----------------
// C[M,N] = A[M,K] @ W[K,N] + bias.
// mode 0: bf16 split-head [B,H,T,HD] output (Q/K/V projections)
// mode 1: fp32 row-major [M,N] output (final projection)
#define GTM 128
#define GTN 128
#define GBK 32
// LDS partition (halves): sA buf = 128*40 = 5120, sB buf = 128*48 = 6144
#define SA_OFF(buf) ((buf) * 5120)
#define SB_OFF(buf) (10240 + (buf) * 6144)

__global__ __launch_bounds__(256) void gemm_bf16_kernel(
    const unsigned short* __restrict__ A,
    const unsigned short* __restrict__ W,
    const float* __restrict__ bias,
    unsigned short* __restrict__ outH,
    float* __restrict__ outF,
    int Kdim, int Ndim, int mode) {
  __shared__ __align__(32) unsigned short smem[22528];  // 45056 B

  const int tid  = threadIdx.x;
  const int lane = tid & 31;
  const int wid  = tid >> 5;
  const int l16  = lane & 15;
  const int hi   = lane >> 4;
  const int m0   = blockIdx.y * GTM;
  const int n0   = blockIdx.x * GTN;
  const int wm   = wid >> 1;           // 0..3
  const int wn   = wid & 1;            // 0..1

  // staging coordinates
  const int ar   = tid >> 2, aseg = tid & 3;   // A rows ar, ar+64; 16B chunk aseg
  const int bk   = tid >> 4, bseg = tid & 15;  // W rows bk, bk+16; 16B chunk bseg

  v8f acc[2][4];
#pragma unroll
  for (int i = 0; i < 2; i++)
#pragma unroll
    for (int j = 0; j < 4; j++) acc[i][j] = v8f_zero();

  float bv[4];
#pragma unroll
  for (int ns = 0; ns < 4; ns++)
    bv[ns] = bias[n0 + wn * 64 + ns * 16 + l16];

  // ---- prologue: stage k-tile 0 into buffer 0 ----
  {
    unsigned short* sA = smem + SA_OFF(0);
    unsigned short* sB = smem + SB_OFF(0);
    async_cp16(A + (size_t)(m0 + ar) * Kdim + aseg * 8,      sA + ar * 40 + aseg * 8);
    async_cp16(A + (size_t)(m0 + ar + 64) * Kdim + aseg * 8, sA + (ar + 64) * 40 + aseg * 8);
    uint4 w0 = *(const uint4*)(W + (size_t)bk * Ndim + n0 + bseg * 8);
    uint4 w1 = *(const uint4*)(W + (size_t)(bk + 16) * Ndim + n0 + bseg * 8);
    const unsigned short* h0 = (const unsigned short*)&w0;
    const unsigned short* h1 = (const unsigned short*)&w1;
#pragma unroll
    for (int e = 0; e < 8; e++) {
      sB[(bseg * 8 + e) * 48 + bk]      = h0[e];
      sB[(bseg * 8 + e) * 48 + bk + 16] = h1[e];
    }
  }
  wait_async0();
  __syncthreads();

  const int nk = Kdim / GBK;
  for (int it = 0; it < nk; ++it) {
    const int cur = it & 1, nxt = cur ^ 1;
    const bool hn = (it + 1) < nk;
    unsigned short* sAc = smem + SA_OFF(cur);
    unsigned short* sBc = smem + SB_OFF(cur);

    uint4 w0 = {}, w1 = {};
    if (hn) {  // issue next tile: A async -> LDS, W -> registers
      const int k0n = (it + 1) * GBK;
      unsigned short* sAn = smem + SA_OFF(nxt);
      async_cp16(A + (size_t)(m0 + ar) * Kdim + k0n + aseg * 8,      sAn + ar * 40 + aseg * 8);
      async_cp16(A + (size_t)(m0 + ar + 64) * Kdim + k0n + aseg * 8, sAn + (ar + 64) * 40 + aseg * 8);
      w0 = *(const uint4*)(W + (size_t)(k0n + bk) * Ndim + n0 + bseg * 8);
      w1 = *(const uint4*)(W + (size_t)(k0n + bk + 16) * Ndim + n0 + bseg * 8);
    }

    // ---- compute on current tile ----
    v16bf af[2];
#pragma unroll
    for (int ms = 0; ms < 2; ms++) {
      int row = wm * 32 + ms * 16 + l16;
      int klo = hi * 8;
      v8bf lo = *(const v8bf*)(sAc + row * 40 + klo);
      v8bf hh = *(const v8bf*)(sAc + row * 40 + klo + 16);
      af[ms] = cat8(lo, hh);
    }
    v16bf bfr[4];
#pragma unroll
    for (int ns = 0; ns < 4; ns++) {
      int n = wn * 64 + ns * 16 + l16;
      bfr[ns] = cat8(*(const v8bf*)(sBc + n * 48 + hi * 8), *(const v8bf*)(sBc + n * 48 + hi * 8 + 16));
    }
#pragma unroll
    for (int ms = 0; ms < 2; ms++)
#pragma unroll
      for (int ns = 0; ns < 4; ns++)
        acc[ms][ns] = __builtin_amdgcn_wmma_f32_16x16x32_bf16(
            false, af[ms], false, bfr[ns], (short)0, acc[ms][ns], false, false);

    if (hn) {  // transpose-store next W tile (loadcnt wait overlaps the WMMAs)
      unsigned short* sBn = smem + SB_OFF(nxt);
      const unsigned short* h0 = (const unsigned short*)&w0;
      const unsigned short* h1 = (const unsigned short*)&w1;
#pragma unroll
      for (int e = 0; e < 8; e++) {
        sBn[(bseg * 8 + e) * 48 + bk]      = h0[e];
        sBn[(bseg * 8 + e) * 48 + bk + 16] = h1[e];
      }
    }
    wait_async0();
    __syncthreads();
  }

  // ---- epilogue ----
  if (mode == 0) {
    // repack through LDS so global writes are coalesced b128
    unsigned short* sO = smem;  // 128 x 128 halves = 32 KB (fits in 45 KB)
#pragma unroll
    for (int ms = 0; ms < 2; ms++)
#pragma unroll
      for (int ns = 0; ns < 4; ns++) {
        float bb = bv[ns];
#pragma unroll
        for (int j = 0; j < 8; j++) {
          int rr = wm * 32 + ms * 16 + hi * 8 + j;
          int cc = wn * 64 + ns * 16 + l16;
          sO[rr * 128 + cc] = f2bf(acc[ms][ns][j] + bb);
        }
      }
    __syncthreads();
#pragma unroll
    for (int q = tid; q < 2048; q += 256) {
      int rr = q >> 4, sg = q & 15;
      int m  = m0 + rr;
      int col = n0 + sg * 8;
      int b = m >> 11;          // m / T_
      int t = m & (T_ - 1);
      int h = col >> 6;
      int d = col & 63;
      *(uint4*)(outH + (((size_t)(b * H_ + h)) * T_ + t) * HD_ + d) =
          *(const uint4*)&sO[rr * 128 + sg * 8];
    }
  } else {
#pragma unroll
    for (int ms = 0; ms < 2; ms++)
#pragma unroll
      for (int ns = 0; ns < 4; ns++) {
        int col = n0 + wn * 64 + ns * 16 + l16;
        float bb = bv[ns];
#pragma unroll
        for (int j = 0; j < 8; j++) {
          int m = m0 + wm * 32 + ms * 16 + hi * 8 + j;
          outF[(size_t)m * Ndim + col] = acc[ms][ns][j] + bb;
        }
      }
  }
}

// ---------------- flash attention (no mask), bf16 WMMA, double-buffered ----------------
__global__ __launch_bounds__(256) void attn_kernel(
    const unsigned short* __restrict__ Q,
    const unsigned short* __restrict__ K,
    const unsigned short* __restrict__ V,
    unsigned short* __restrict__ Y) {
  __shared__ __align__(32) unsigned short sK[2][32][80];    // [key][d]
  __shared__ __align__(32) unsigned short sVT[2][64][48];   // [d][key]
  __shared__ __align__(16) unsigned short sP[8][16][32];    // per-wave P scratch

  const int tid  = threadIdx.x;
  const int lane = tid & 31;
  const int wid  = tid >> 5;
  const int l16  = lane & 15;
  const int hi   = lane >> 4;
  const int bh   = blockIdx.x;
  const int q0   = blockIdx.y * 128;
  const int b    = bh >> 4;
  const int h    = bh & 15;

  const unsigned short* Qh = Q + (size_t)bh * T_ * HD_;
  const unsigned short* Kh = K + (size_t)bh * T_ * HD_;
  const unsigned short* Vh = V + (size_t)bh * T_ * HD_;

  const int r   = tid >> 3;   // staging row 0..31
  const int seg = tid & 7;    // 16B chunk

  // Q fragments (16 rows, contraction over HD=64 -> 2 frags)
  v16bf qa[2];
  {
    int row = q0 + wid * 16 + l16;
    int klo = hi * 8;
#pragma unroll
    for (int ks = 0; ks < 2; ks++) {
      v8bf lo = *(const v8bf*)(Qh + (size_t)row * HD_ + ks * 32 + klo);
      v8bf hh = *(const v8bf*)(Qh + (size_t)row * HD_ + ks * 32 + klo + 16);
      qa[ks] = cat8(lo, hh);
    }
  }

  v8f o[4];
#pragma unroll
  for (int ns = 0; ns < 4; ns++) o[ns] = v8f_zero();
  float mrow[8], lrow[8];
#pragma unroll
  for (int j = 0; j < 8; j++) { mrow[j] = -3.0e38f; lrow[j] = 0.f; }

  // prologue: stage tile 0 into buffer 0 (K via TDM if available, else async)
#if HAS_TDM
  if (wid == 0)
    tdm_load_k_tile(Kh, (unsigned)(unsigned long long)&sK[0][0][0]);
#else
  async_cp16(Kh + (size_t)r * HD_ + seg * 8, &sK[0][r][seg * 8]);
#endif
  {
    uint4 w = *(const uint4*)(Vh + (size_t)r * HD_ + seg * 8);
    const unsigned short* hh = (const unsigned short*)&w;
#pragma unroll
    for (int e = 0; e < 8; e++) sVT[0][seg * 8 + e][r] = hh[e];
  }
#if HAS_TDM
  if (wid == 0) wait_tensor0();
#endif
  wait_async0();
  __syncthreads();

  const int NT = T_ / 32;  // 64 key tiles
  for (int it = 0; it < NT; ++it) {
    const int cur = it & 1, nxt = cur ^ 1;
    const bool hn = (it + 1) < NT;
    uint4 vreg = {};
    if (hn) {  // prefetch next tile: K -> LDS (TDM/async), V -> registers
      int kt = (it + 1) * 32;
#if HAS_TDM
      if (wid == 0)
        tdm_load_k_tile(Kh + (size_t)kt * HD_,
                        (unsigned)(unsigned long long)&sK[nxt][0][0]);
#else
      async_cp16(Kh + (size_t)(kt + r) * HD_ + seg * 8, &sK[nxt][r][seg * 8]);
#endif
      vreg = *(const uint4*)(Vh + (size_t)(kt + r) * HD_ + seg * 8);
    }

    // scores S(16x32) = Q @ K_tile^T
    v8f s[2];
    s[0] = v8f_zero();
    s[1] = v8f_zero();
#pragma unroll
    for (int ns = 0; ns < 2; ns++)
#pragma unroll
      for (int ks = 0; ks < 2; ks++) {
        v16bf kbf = cat8(*(const v8bf*)&sK[cur][ns * 16 + l16][ks * 32 + hi * 8], *(const v8bf*)&sK[cur][ns * 16 + l16][ks * 32 + hi * 8 + 16]);
        s[ns] = __builtin_amdgcn_wmma_f32_16x16x32_bf16(
            false, qa[ks], false, kbf, (short)0, s[ns], false, false);
      }

    // online softmax (scale = 1/sqrt(64) = 0.125)
#pragma unroll
    for (int j = 0; j < 8; j++) {
      float s0 = s[0][j] * 0.125f;
      float s1 = s[1][j] * 0.125f;
      float mx = fmaxf(s0, s1);
#pragma unroll
      for (int off = 1; off < 16; off <<= 1) mx = fmaxf(mx, __shfl_xor(mx, off, 32));
      float mnew = fmaxf(mrow[j], mx);
      float a  = __expf(mrow[j] - mnew);
      float p0 = __expf(s0 - mnew);
      float p1 = __expf(s1 - mnew);
      float rs = p0 + p1;
#pragma unroll
      for (int off = 1; off < 16; off <<= 1) rs += __shfl_xor(rs, off, 32);
      mrow[j] = mnew;
      lrow[j] = lrow[j] * a + rs;
      sP[wid][hi * 8 + j][l16]      = f2bf(p0);
      sP[wid][hi * 8 + j][16 + l16] = f2bf(p1);
#pragma unroll
      for (int ns = 0; ns < 4; ns++) o[ns][j] *= a;
    }
    asm volatile("s_wait_dscnt 0" ::: "memory");  // same-wave LDS RAW on sP

    // O += P @ V_tile
    v16bf pa;
    {
      int klo = hi * 8;
      v8bf lo = *(const v8bf*)&sP[wid][l16][klo];
      v8bf hh = *(const v8bf*)&sP[wid][l16][klo + 16];
      pa = cat8(lo, hh);
    }
#pragma unroll
    for (int ns = 0; ns < 4; ns++) {
      v16bf vbf = cat8(*(const v8bf*)&sVT[cur][ns * 16 + l16][hi * 8], *(const v8bf*)&sVT[cur][ns * 16 + l16][hi * 8 + 16]);
      o[ns] = __builtin_amdgcn_wmma_f32_16x16x32_bf16(
          false, pa, false, vbf, (short)0, o[ns], false, false);
    }

    if (hn) {  // transpose-store next V tile after compute
      const unsigned short* hh = (const unsigned short*)&vreg;
#pragma unroll
      for (int e = 0; e < 8; e++) sVT[nxt][seg * 8 + e][r] = hh[e];
    }
#if HAS_TDM
    if (wid == 0) wait_tensor0();
#endif
    wait_async0();
    __syncthreads();
  }

  // finalize: divide by denominator, store bf16 y[b][t][h*64+d]
#pragma unroll
  for (int j = 0; j < 8; j++) {
    float inv = 1.0f / lrow[j];
    int t = q0 + wid * 16 + hi * 8 + j;
#pragma unroll
    for (int ns = 0; ns < 4; ns++) {
      int c = h * 64 + ns * 16 + l16;
      Y[((size_t)b * T_ + t) * C_ + c] = f2bf(o[ns][j] * inv);
    }
  }
}

// ---------------- host launch ----------------
extern "C" void kernel_launch(void* const* d_in, const int* in_sizes, int n_in,
                              void* d_out, int out_size, void* d_ws, size_t ws_size,
                              hipStream_t stream) {
  (void)in_sizes; (void)n_in; (void)out_size; (void)ws_size;
  const float* x  = (const float*)d_in[0];
  const float* Wq = (const float*)d_in[1];
  const float* bq = (const float*)d_in[2];
  const float* Wk = (const float*)d_in[3];
  const float* bk = (const float*)d_in[4];
  const float* Wv = (const float*)d_in[5];
  const float* bv = (const float*)d_in[6];
  const float* Wo = (const float*)d_in[7];
  const float* bo = (const float*)d_in[8];

  char* ws = (char*)d_ws;
  const size_t MB = 1024 * 1024;
  unsigned short* xb  = (unsigned short*)(ws);
  unsigned short* wqb = (unsigned short*)(ws +  8 * MB);
  unsigned short* wkb = (unsigned short*)(ws + 10 * MB);
  unsigned short* wvb = (unsigned short*)(ws + 12 * MB);
  unsigned short* wob = (unsigned short*)(ws + 14 * MB);
  unsigned short* qb  = (unsigned short*)(ws + 16 * MB);
  unsigned short* kb  = (unsigned short*)(ws + 24 * MB);
  unsigned short* vb  = (unsigned short*)(ws + 32 * MB);
  unsigned short* yb  = (unsigned short*)(ws + 40 * MB);

  cvt_f32_bf16<<<1024, 256, 0, stream>>>(x,  xb,  (M_ * C_) / 4);
  cvt_f32_bf16<<<512,  256, 0, stream>>>(Wq, wqb, (C_ * C_) / 4);
  cvt_f32_bf16<<<512,  256, 0, stream>>>(Wk, wkb, (C_ * C_) / 4);
  cvt_f32_bf16<<<512,  256, 0, stream>>>(Wv, wvb, (C_ * C_) / 4);
  cvt_f32_bf16<<<512,  256, 0, stream>>>(Wo, wob, (C_ * C_) / 4);

  dim3 ggrid(C_ / GTN, M_ / GTM);  // (8, 32)
  gemm_bf16_kernel<<<ggrid, 256, 0, stream>>>(xb, wqb, bq, qb, nullptr, C_, C_, 0);
  gemm_bf16_kernel<<<ggrid, 256, 0, stream>>>(xb, wkb, bk, kb, nullptr, C_, C_, 0);
  gemm_bf16_kernel<<<ggrid, 256, 0, stream>>>(xb, wvb, bv, vb, nullptr, C_, C_, 0);

  attn_kernel<<<dim3(B_ * H_, T_ / 128), 256, 0, stream>>>(qb, kb, vb, yb);

  gemm_bf16_kernel<<<ggrid, 256, 0, stream>>>(yb, wob, bo, nullptr, (float*)d_out,
                                              C_, C_, 1);
}
